// DeformableConv1D_22393959481995
// MI455X (gfx1250) — matrix operand order repaired and checked by reading, not compiled, not run on hardware
//
#include <hip/hip_runtime.h>
#include <hip/hip_bf16.h>

typedef __attribute__((ext_vector_type(16))) __bf16          v16bf;
typedef __attribute__((ext_vector_type(8)))  float           v8f;

#define BATCH  4
#define CIN    256
#define COUT   256
#define LEN    8192
#define KW     7
#define KDIM   (CIN * KW)     // 1792
#define KSTEPS (KDIM / 32)    // 56
#define MTILE  64

struct alignas(16) Meta { float wa, wb; int x0, x1; };

// ---------------------------------------------------------------------------
// Kernel 1: fp32 weight (COUT, CIN, KW) -> bf16 in WMMA B-fragment layout:
//   wbf[ ((nt*KSTEPS + kt)*32 + lane)*16 + j ]
// B (32x16) layout: lanes 0-15 hold K=0..15 (elem j -> K=j), lanes 16-31 hold
// K=16..31; column N = lane & 15.
// ---------------------------------------------------------------------------
__global__ __launch_bounds__(256) void wconv_kernel(const float* __restrict__ w,
                                                    __bf16* __restrict__ wbf) {
  int t = blockIdx.x * 256 + threadIdx.x;
  if (t >= COUT * KDIM) return;
  int o  = t / KDIM;
  int kd = t - o * KDIM;
  int ci = kd / KW;
  int kk = kd - ci * KW;
  float v = w[(o * CIN + ci) * KW + kk];
  int nt   = o >> 4;
  int col  = o & 15;
  int kt   = kd >> 5;
  int Kloc = kd & 31;
  int lane = col + ((Kloc >= 16) ? 16 : 0);
  int j    = Kloc & 15;
  wbf[(((nt * KSTEPS + kt) * 32 + lane) << 4) + j] = (__bf16)v;   // native cvt
}

// ---------------------------------------------------------------------------
// Kernel 2: fused offset-conv + deformable gather/lerp + bf16 WMMA GEMM.
// Block tile: M=64 (rows of b*out_len), N=256 (all Cout). 8 waves:
//   wave = (mslice 0..3) x (nhalf 0..1); each wave -> 16 rows x 128 cols.
// ---------------------------------------------------------------------------
__global__ __launch_bounds__(256) void deform_gemm_kernel(
    const float* __restrict__ x, const __bf16* __restrict__ wbf,
    const float* __restrict__ bias, const float* __restrict__ offw,
    const float* __restrict__ offb, float* __restrict__ out) {

  __shared__ Meta  metaLds[MTILE * KW];      // 7168 B
  __shared__ float outStage[8][16][17];      // 8704 B (pad 17 kills bank conflicts)

  const int tid = threadIdx.x;
  const int m0  = blockIdx.x * MTILE;        // flat row = b*LEN + l
  const int b   = m0 >> 13;
  const int l0  = m0 & (LEN - 1);
  const float* xb = x + (size_t)b * CIN * LEN;

  // ---- prologue: offset conv + sampling meta for this block's 64 rows ----
  const bool interior = (l0 >= 3) && (l0 + MTILE + 3 <= LEN);
  for (int item = tid; item < MTILE * KW; item += 256) {
    int row = item / KW;
    int ko  = item - row * KW;
    int l   = l0 + row;
    float acc = offb[ko];
    const float* wr = offw + ko * CIN * KW;
    if (interior) {
      // fast path: no boundary predication -> mergeable consecutive loads
      const float* xp = xb + (l - 3);
      for (int ci = 0; ci < CIN; ++ci) {
        const float* xr = xp + ci * LEN;
        #pragma unroll
        for (int kk = 0; kk < KW; ++kk)
          acc = fmaf(xr[kk], wr[ci * KW + kk], acc);
      }
    } else {
      for (int ci = 0; ci < CIN; ++ci) {
        const float* xr = xb + ci * LEN;
        #pragma unroll
        for (int kk = 0; kk < KW; ++kk) {
          int pos = l + kk - 3;               // stride 1, pad 3, dil 1
          float xv = (pos >= 0 && pos < LEN) ? xr[pos] : 0.0f;
          acc = fmaf(xv, wr[ci * KW + kk], acc);
        }
      }
    }
    float loc = (float)(l + ko) + acc;        // l + (k-3) + PAD + offset
    int x0 = (int)floorf(loc);
    int x1 = x0 + 1;
    x0 = min(max(x0, 0), LEN - 1);
    x1 = min(max(x1, 0), LEN - 1);
    Meta m;
    m.wa = (float)x1 - loc;                   // weights use clamped indices
    m.wb = loc - (float)x0;
    m.x0 = x0; m.x1 = x1;
    metaLds[item] = m;                        // ds_store_b128
  }
  __syncthreads();

  // ---- main GEMM loop -----------------------------------------------------
  const int wave   = tid >> 5;
  const int lane   = tid & 31;
  const int msl    = wave >> 1;               // M slice 0..3
  const int nhalf  = wave & 1;                // N half  0..1
  const int laneLo = lane & 15;
  const int laneHi = lane >> 4;
  const int rowl   = msl * 16 + laneLo;       // block-local row (= A-matrix M row)
  const int khi    = laneHi * 8;              // upper lanes: A K+16 (via 2*khi), C/D M+8

  v8f c[8];
  #pragma unroll
  for (int s = 0; s < 8; ++s) {
    v8f z = {0.f, 0.f, 0.f, 0.f, 0.f, 0.f, 0.f, 0.f};
    c[s] = z;
  }

  const v16bf* bfrag = ((const v16bf*)wbf) +
                       ((size_t)(nhalf * 8) * KSTEPS) * 32 + lane;

  for (int kt = 0; kt < KSTEPS; ++kt) {
    // Build A fragment (16x32 bf16) in contiguous K order, same as wbf:
    //   lanes 0-15:  elem j -> K=j
    //   lanes 16-31: elem j -> K=j+16
    v16bf a;
    #pragma unroll
    for (int j = 0; j < 16; ++j) {
      int Kloc = j + 2 * khi;
      int kk = kt * 32 + Kloc;                // flat K index = ci*KW + k
      int ci = kk / KW;
      int k  = kk - ci * KW;
      Meta m = metaLds[rowl * KW + k];        // ds_load_b128
      const float* xr = xb + (ci << 13);
      float v = m.wa * xr[m.x0] + m.wb * xr[m.x1];   // gather + lerp
      a[j] = (__bf16)v;                       // native f32->bf16 cvt
    }

    #pragma unroll
    for (int s = 0; s < 8; ++s) {
      v16bf bb = bfrag[((size_t)s * KSTEPS + kt) * 32];   // 32B contiguous
      c[s] = __builtin_amdgcn_wmma_f32_16x16x32_bf16(
                 false, a, false, bb, (short)0, c[s], false, false);
    }
  }

  // ---- epilogue: LDS transpose per wave so stores are contiguous in l ----
  const int lbase = l0 + msl * 16;
  float (*st)[17] = outStage[wave];
  for (int s = 0; s < 8; ++s) {
    // C/D layout: VGPR r -> (M = r + 8*laneHi, N = laneLo)
    #pragma unroll
    for (int r = 0; r < 8; ++r)
      st[laneLo][r + khi] = c[s][r];
    __asm__ volatile("s_wait_dscnt 0" ::: "memory");
    int obase = nhalf * 128 + s * 16;
    #pragma unroll
    for (int t2 = 0; t2 < 8; ++t2) {
      int ol = t2 * 2 + laneHi;
      float v = st[ol][laneLo] + bias[obase + ol];
      out[((size_t)(b * COUT + obase + ol) << 13) + lbase + laneLo] = v;
    }
    __asm__ volatile("s_wait_dscnt 0" ::: "memory");
  }
}

// ---------------------------------------------------------------------------
extern "C" void kernel_launch(void* const* d_in, const int* in_sizes, int n_in,
                              void* d_out, int out_size, void* d_ws, size_t ws_size,
                              hipStream_t stream) {
  const float* x    = (const float*)d_in[0];   // (4, 256, 8192)
  const float* w    = (const float*)d_in[1];   // (256, 256, 7)
  const float* bias = (const float*)d_in[2];   // (256,)
  const float* offw = (const float*)d_in[3];   // (7, 256, 7)
  const float* offb = (const float*)d_in[4];   // (7,)
  float* out = (float*)d_out;                  // (4, 256, 8192)

  __bf16* wbf = (__bf16*)d_ws;                 // needs COUT*KDIM*2 = 917504 B

  wconv_kernel<<<(COUT * KDIM + 255) / 256, 256, 0, stream>>>(w, wbf);
  deform_gemm_kernel<<<(BATCH * LEN) / MTILE, 256, 0, stream>>>(
      x, wbf, bias, offw, offb, out);
}
